// VanillaRNN_28286654611945
// MI455X (gfx1250) — matrix operand order repaired and checked by reading, not compiled, not run on hardware
//
#include <hip/hip_runtime.h>
#include <hip/hip_bf16.h>
#include <math.h>

// Problem constants (from reference)
#define BB   128
#define TT   2048
#define ENC  64
#define HH   256

typedef __attribute__((ext_vector_type(16))) _Float16 v16h;
typedef __attribute__((ext_vector_type(8)))  _Float16 v8h;
typedef __attribute__((ext_vector_type(8)))  float    v8f;

static constexpr int    LDW       = HH + 8;              // padded LDS row
static constexpr int    TILE_ELTS = 16 * HH;             // 4096 f16 per (tile,t)
// P layout (f16): [run*8+mb][t][wave(16)][lane(32)][i(8)]
//   value (m,n) of tile lives at: wave = n>>4 ; lane = (n&15) + ((m&8)?16:0) ; i = m&7
// This IS the WMMA C/D fragment layout, so scan lanes load/store v8h contiguously.

// ---- fast tanh on the serial path: v_exp_f32 + v_rcp_f32 --------------------
// tanh(x) = 1 - 2/(exp(2x)+1); saturates correctly for large |x|, NaN propagates.
__device__ inline float fast_tanh(float x) {
    float e = __builtin_amdgcn_exp2f(x * 2.885390081777927f);  // exp(2x)=2^(2x*log2e)
    return 1.0f - 2.0f * __builtin_amdgcn_rcpf(e + 1.0f);
}

// ---- fragment loaders -------------------------------------------------------

// A-matrix fragment (16x32 f16) from LDS h-tile [16][LDW].
__device__ inline v16h lds_load_A(const _Float16* hsh, int kb, int lane) {
    int m  = lane & 15;
    int ko = kb * 32 + ((lane & 16) ? 8 : 0);
    const _Float16* p = hsh + m * LDW + ko;
    v8h lo = *(const v8h*)p;
    v8h hi = *(const v8h*)(p + 16);
    return __builtin_shufflevector(lo, hi, 0,1,2,3,4,5,6,7,8,9,10,11,12,13,14,15);
}

// B-matrix fragment (32x16 f16) for C = h * W^T, i.e. B[k][n] = W[n][k].
__device__ inline v16h glb_load_B(const float* __restrict__ W, int nbase, int kb, int lane) {
    int n  = nbase + (lane & 15);
    int ko = kb * 32 + ((lane & 16) ? 8 : 0);
    const float* p = W + (size_t)n * HH + ko;
    v16h b;
#pragma unroll
    for (int j = 0; j < 16; ++j) b[j] = (_Float16)p[j < 8 ? j : j + 8];
    return b;
}

__device__ inline v8f wmma_f16(v16h a, v16h b, v8f c) {
    return __builtin_amdgcn_wmma_f32_16x16x32_f16(false, a, false, b, (short)0, c, false, false);
}

// ---- kernel 0: fold encoder + W_ih0 into a 2x256 affine map -----------------
__global__ void fuse_enc_kernel(const float* __restrict__ encW,   // [ENC][2]
                                const float* __restrict__ encB,   // [ENC]
                                const float* __restrict__ Wih0,   // [H][ENC]
                                const float* __restrict__ bih0,   // [H]
                                float* __restrict__ Wc)           // Wc0[H],Wc1[H],bc[H]
{
    int n = threadIdx.x;
    float wc0 = 0.f, wc1 = 0.f, bac = 0.f;
    for (int e = 0; e < ENC; ++e) {
        float w = Wih0[n * ENC + e];
        wc0 += w * encW[e * 2 + 0];
        wc1 += w * encW[e * 2 + 1];
        bac += w * encB[e];
    }
    Wc[n]          = wc0;
    Wc[HH + n]     = wc1;
    Wc[2 * HH + n] = bac + bih0[n];
}

// ---- kernel 1: pre0 directly in fragment-swizzled layout --------------------
__global__ __launch_bounds__(512)
void prep_kernel(const float* __restrict__ X,    // [B][T][2]
                 const float* __restrict__ Wc,
                 _Float16* __restrict__ P)
{
    const int idx  = blockIdx.x;          // (r*8+mb)*T + t
    const int t    = idx & (TT - 1);
    const int rm   = idx >> 11;
    const int r    = rm >> 3;
    const int mb   = rm & 7;
    const int tid  = threadIdx.x;
    const int lane = tid & 31;
    const int wv   = tid >> 5;
    const int n    = wv * 16 + (lane & 15);
    const int mhi  = (lane & 16) ? 8 : 0;

    const float w0 = Wc[n], w1 = Wc[HH + n], bc = Wc[2 * HH + n];
    v8h out;
#pragma unroll
    for (int i = 0; i < 8; ++i) {
        int b = mb * 16 + mhi + i;
        float x0 = X[((size_t)b * TT + t) * 2 + 0];
        float x1 = X[((size_t)b * TT + t) * 2 + 1];
        if (r) x0 = -x0;
        out[i] = (_Float16)(x0 * w0 + x1 * w1 + bc);
    }
    *(v8h*)(P + (size_t)idx * TILE_ELTS + wv * 256 + lane * 8) = out;
}

// ---- scan kernel: persistent over T, one block = (run, 16-row batch tile) ---
// FUSE=1: layer0 + fused layer-1 input GEMM written in place.
// FUSE=0: layer1; writes h_t (f16) in place.  All P traffic is v8h (b128).
template <int FUSE>
__global__ __launch_bounds__(512)
void scan_kernel(_Float16* P,                       // swizzled [rm][t][frag]
                 const float* __restrict__ Whh,     // [H][H]
                 const float* __restrict__ bhh,     // [H]
                 const float* __restrict__ Wnext,   // [H][H] (FUSE only)
                 const float* __restrict__ bnext,   // [H]    (FUSE only)
                 const float* __restrict__ h0)      // [B][H] initial hidden
{
    const int blk  = blockIdx.x;       // 0..15
    const int run  = blk >> 3;
    const int mb   = blk & 7;
    const int tid  = threadIdx.x;
    const int lane = tid & 31;
    const int wave = tid >> 5;         // owns N columns [16*wave, 16*wave+16)
    const int nbase = wave * 16;
    const int b0    = mb * 16;

    __shared__ __align__(16) _Float16 hsh[16 * LDW];

    // persistent weight fragments in VGPRs
    v16h Bh[8];
#pragma unroll
    for (int kb = 0; kb < 8; ++kb) Bh[kb] = glb_load_B(Whh, nbase, kb, lane);
    v16h Bn[8];
    if (FUSE) {
#pragma unroll
        for (int kb = 0; kb < 8; ++kb) Bn[kb] = glb_load_B(Wnext, nbase, kb, lane);
    }

    const int   nlane = nbase + (lane & 15);
    const int   mhi   = (lane & 16) ? 8 : 0;
    const float bias1 = bhh[nlane];
    const float bias2 = FUSE ? bnext[nlane] : 0.f;

    // init hidden state tile into LDS (f32 -> f16)
    for (int i = tid; i < 16 * HH; i += 512) {
        int m = i >> 8, n = i & (HH - 1);
        hsh[m * LDW + n] = (_Float16)h0[(size_t)(b0 + m) * HH + n];
    }
    __syncthreads();

    // per-lane streaming pointer into swizzled P; advances 8KB per step
    _Float16* Pt = P + (size_t)(run * 8 + mb) * TT * TILE_ELTS
                     + wave * 256 + lane * 8;

    v8h pre = *(const v8h*)Pt;            // software-pipelined pre[t]
    for (int t = 0; t < TT; ++t) {
        // issue pre[t+1] load early to overlap with the WMMA chain
        v8h preN = *(const v8h*)(Pt + ((t + 1 < TT) ? TILE_ELTS : 0));

        // phase 1: h_new = tanh(pre[t] + h_old * Whh^T + bhh)
        v16h A[8];
#pragma unroll
        for (int kb = 0; kb < 8; ++kb) A[kb] = lds_load_A(hsh, kb, lane);

        v8f c;
#pragma unroll
        for (int i = 0; i < 8; ++i) c[i] = (float)pre[i] + bias1;
#pragma unroll
        for (int kb = 0; kb < 8; ++kb) c = wmma_f16(A[kb], Bh[kb], c);

        v8h hn;
#pragma unroll
        for (int i = 0; i < 8; ++i) hn[i] = (_Float16)fast_tanh(c[i]);

        __syncthreads();   // all waves done reading h_old
#pragma unroll
        for (int i = 0; i < 8; ++i) hsh[(mhi + i) * LDW + nlane] = hn[i];
        __syncthreads();   // h_new fully visible

        if (FUSE) {
            // phase 2: pre_next[t] = h_new * Wnext^T + bnext (in place, one b128)
            v16h A2[8];
#pragma unroll
            for (int kb = 0; kb < 8; ++kb) A2[kb] = lds_load_A(hsh, kb, lane);
            v8f c2;
#pragma unroll
            for (int i = 0; i < 8; ++i) c2[i] = bias2;
#pragma unroll
            for (int kb = 0; kb < 8; ++kb) c2 = wmma_f16(A2[kb], Bn[kb], c2);
            v8h o;
#pragma unroll
            for (int i = 0; i < 8; ++i) o[i] = (_Float16)c2[i];
            *(v8h*)Pt = o;
        } else {
            *(v8h*)Pt = hn;               // store h_t for the decoder
        }
        Pt += TILE_ELTS;
        pre = preN;
    }
}

// ---- final: decoder + symmetric flip-average, one wave per (b,t) ------------
__global__ __launch_bounds__(32)
void decode_kernel(const _Float16* __restrict__ P,   // swizzled y1, both runs
                   const float* __restrict__ decW,   // [2][H]
                   const float* __restrict__ decB,   // [2]
                   float* __restrict__ out)          // [B][T][2]
{
    const size_t bt = blockIdx.x;            // b*T + t
    const int b    = (int)(bt >> 11);
    const int t    = (int)(bt & (TT - 1));
    const int mb   = b >> 4;
    const int m    = b & 15;
    const int lane = threadIdx.x;

    const size_t baseA = ((size_t)(0 + mb) * TT + t) * TILE_ELTS;
    const size_t baseB = ((size_t)(8 + mb) * TT + t) * TILE_ELTS;

    const int wv   = lane & 15;              // n-group
    const int half = lane >> 4;              // low/high 8 of the 16-col group
    const int mlo  = m & 7;
    const int msel = (m & 8) ? 16 : 0;

    float dA0 = 0.f, dA1 = 0.f, dB0 = 0.f, dB1 = 0.f;
#pragma unroll
    for (int j = 0; j < 8; ++j) {
        int n   = wv * 16 + half * 8 + j;
        int off = wv * 256 + (half * 8 + j + msel) * 8 + mlo;
        float w0 = decW[n];
        float w1 = decW[HH + n];
        float a  = (float)P[baseA + off];
        float bb = (float)P[baseB + off];
        dA0 += a * w0;  dA1 += a * w1;
        dB0 += bb * w0; dB1 += bb * w1;
    }
    for (int off = 16; off > 0; off >>= 1) {
        dA0 += __shfl_down(dA0, off);
        dA1 += __shfl_down(dA1, off);
        dB0 += __shfl_down(dB0, off);
        dB1 += __shfl_down(dB1, off);
    }
    if (lane == 0) {
        out[bt * 2 + 0] = 0.5f * ((dA0 + decB[0]) + (dB1 + decB[1]));
        out[bt * 2 + 1] = 0.5f * ((dA1 + decB[1]) + (dB0 + decB[0]));
    }
}

// ---- launcher ---------------------------------------------------------------
extern "C" void kernel_launch(void* const* d_in, const int* in_sizes, int n_in,
                              void* d_out, int out_size, void* d_ws, size_t ws_size,
                              hipStream_t stream) {
    const float* X     = (const float*)d_in[0];
    const float* hid   = (const float*)d_in[1];   // [2][B][H]
    const float* encW  = (const float*)d_in[2];
    const float* encB  = (const float*)d_in[3];
    const float* Wih0  = (const float*)d_in[4];
    const float* Whh0  = (const float*)d_in[5];
    const float* bih0  = (const float*)d_in[6];
    const float* bhh0  = (const float*)d_in[7];
    const float* Wih1  = (const float*)d_in[8];
    const float* Whh1  = (const float*)d_in[9];
    const float* bih1  = (const float*)d_in[10];
    const float* bhh1  = (const float*)d_in[11];
    const float* decW  = (const float*)d_in[12];
    const float* decB  = (const float*)d_in[13];
    float* out = (float*)d_out;

    // workspace: swizzled P (f16, 2 runs) then folded encoder map
    _Float16* P  = (_Float16*)d_ws;
    float*    Wc = (float*)((char*)d_ws +
                   (size_t)16 * TT * TILE_ELTS * sizeof(_Float16));

    fuse_enc_kernel<<<1, 256, 0, stream>>>(encW, encB, Wih0, bih0, Wc);
    prep_kernel<<<16 * TT, 512, 0, stream>>>(X, Wc, P);
    scan_kernel<1><<<16, 512, 0, stream>>>(P, Whh0, bhh0, Wih1, bih1, hid);
    scan_kernel<0><<<16, 512, 0, stream>>>(P, Whh1, bhh1, nullptr, nullptr,
                                           hid + (size_t)BB * HH);
    decode_kernel<<<BB * TT, 32, 0, stream>>>(P, decW, decB, out);
}
